// VQVAE_11209864642758
// MI455X (gfx1250) — matrix operand order repaired and checked by reading, not compiled, not run on hardware
//
#include <hip/hip_runtime.h>
#include <stdint.h>

// ---------------------------------------------------------------------------
// VQ-VAE nearest-codebook lookup for MI455X (gfx1250, wave32, WMMA).
//   N = 131072 rows, K = 4096 codes, D = 64.
//   argmin_k ( |c_k|^2 + BIAS - 2 * x . c_k )   (row-constant |x|^2 dropped,
//   BIAS makes keys positive so uint bit-order == float order)
//   cross term via v_wmma_f32_16x16x32_bf16, |c|^2 exact in f32.
//   Packed (dist|code) keys -> single v_min_u32 argmin per element.
//   Software-pipelined inner loop + double-buffered LDS codebook chunks.
// ---------------------------------------------------------------------------

typedef __attribute__((ext_vector_type(16))) __bf16 v16bf;
typedef __attribute__((ext_vector_type(8)))  float  v8f;

union BF16Frag {
  v16bf v;
  uint4 q[2];
};

struct BPipe {
  BF16Frag b0, b1;   // B fragments for dims 0..31 and 32..63
  float cn;          // |c|^2 + BIAS for this lane's code column
};

__device__ __forceinline__ unsigned int f2bf(float f) {
  unsigned int u = __float_as_uint(f);
  unsigned int r = 0x7FFFu + ((u >> 16) & 1u);   // round-to-nearest-even
  return (u + r) >> 16;
}

#define TILE_M   128     // rows per block (8 waves x 16)
#define TILE_KC  128     // codes per LDS chunk
#define DIM      64
#define DIST_BIAS 1024.0f  // makes all keys positive (|2 x.c| << 1024 here)
#define IDX_MASK  0xFFFu   // K = 4096 codes -> 12 index bits in the key

// ---------------------------------------------------------------------------
// Setup: codebook f32 -> bf16 copy + biased per-code squared norms.
// ---------------------------------------------------------------------------
__global__ __launch_bounds__(256)
void vq_setup_kernel(const float* __restrict__ CB,
                     unsigned short* __restrict__ CBh,
                     float* __restrict__ CN, int K) {
  int k = blockIdx.x * blockDim.x + threadIdx.x;
  if (k >= K) return;
  const float* row = CB + (size_t)k * DIM;
  unsigned short* dst = CBh + (size_t)k * DIM;
  float s = 0.0f;
#pragma unroll
  for (int d = 0; d < DIM; d += 4) {
    float4 v = *(const float4*)(row + d);
    s += v.x * v.x + v.y * v.y + v.z * v.z + v.w * v.w;
    uint2 p;
    p.x = f2bf(v.x) | (f2bf(v.y) << 16);
    p.y = f2bf(v.z) | (f2bf(v.w) << 16);
    *(uint2*)(dst + d) = p;
  }
  CN[k] = s + DIST_BIAS;
}

// ---------------------------------------------------------------------------
// Helpers for the pipelined inner loop.
// ---------------------------------------------------------------------------
__device__ __forceinline__ void load_b(const unsigned short* cb,
                                       const float* cna,
                                       int sub, int nlo, int hb, BPipe& p) {
  // B fragment (32x16 bf16): lane holds column nlo, K = {hb*8 + 0..7, 16 + hb*8 + 0..7}
  const unsigned short* br = &cb[(sub * 16 + nlo) * DIM + hb * 8];
  p.b0.q[0] = ((const uint4*)br)[0];
  p.b0.q[1] = ((const uint4*)br)[2];
  p.b1.q[0] = ((const uint4*)(br + 32))[0];
  p.b1.q[1] = ((const uint4*)(br + 32))[2];
  p.cn = cna[sub * 16 + nlo];
}

__device__ __forceinline__ v8f wmma2(const BF16Frag& a0, const BF16Frag& a1,
                                     const BPipe& p) {
  v8f c = {};
  c = __builtin_amdgcn_wmma_f32_16x16x32_bf16(
      false, a0.v, false, p.b0.v, (short)0, c, false, false);
  c = __builtin_amdgcn_wmma_f32_16x16x32_bf16(
      false, a1.v, false, p.b1.v, (short)0, c, false, false);
  return c;
}

// Packed-key argmin: dist is positive -> float order == uint bit order.
// key = (bits(dist) & ~IDX_MASK) | code ; ties resolve to smallest code.
// Per slot: v_fma + v_and_or_b32 + v_min_u32 (3 VALU, co-execs with WMMA).
__device__ __forceinline__ void min_update(const v8f& acc, float cn,
                                           unsigned code, unsigned* mink) {
#pragma unroll
  for (int j = 0; j < 8; ++j) {
    float d = fmaf(-2.0f, acc[j], cn);
    unsigned key = (__float_as_uint(d) & ~IDX_MASK) | code;
    mink[j] = mink[j] < key ? mink[j] : key;
  }
}

// ---------------------------------------------------------------------------
// Main kernel: WMMA distance matrix + fused argmin + gather.
// ---------------------------------------------------------------------------
__global__ __launch_bounds__(256)
void vq_main_kernel(const float* __restrict__ X,            // [N, 64] f32
                    const float* __restrict__ CB,           // [K, 64] f32
                    const unsigned short* __restrict__ CBh, // [K, 64] bf16
                    const float* __restrict__ CN,           // [K] biased norms
                    float* __restrict__ out,                // [2, N, 64] f32
                    int nrows, int K) {
  __shared__ alignas(16) unsigned short lds_x[TILE_M * DIM];       // 16 KB
  __shared__ alignas(16) unsigned short lds_cb[2][TILE_KC * DIM];  // 2x16 KB
  __shared__ float lds_cn[2][TILE_KC];
  __shared__ int   lds_idx[TILE_M];

  const int t    = threadIdx.x;       // 0..255
  const int wave = t >> 5;            // 0..7
  const int lane = t & 31;            // wave32
  const int nlo  = lane & 15;         // column within 16x16 tile
  const int hb   = lane >> 4;         // half-wave selector
  const int block_row = blockIdx.x * TILE_M;

  // ---- Stage X tile (128x64 f32 -> bf16 LDS), coalesced float4 loads ----
  {
    const float4* xp = (const float4*)(X + (size_t)block_row * DIM);
#pragma unroll
    for (int j = 0; j < 8; ++j) {
      int g = t + 256 * j;            // float4 group, 0..2047
      float4 v = xp[g];
      uint2 p;
      p.x = f2bf(v.x) | (f2bf(v.y) << 16);
      p.y = f2bf(v.z) | (f2bf(v.w) << 16);
      *(uint2*)&lds_x[g * 4] = p;
    }
  }

  // ---- Stage first codebook chunk into buffer 0 ----
  {
    const uint4* src = (const uint4*)CBh;
    uint4* dst = (uint4*)lds_cb[0];
#pragma unroll
    for (int j = 0; j < 4; ++j) dst[t + 256 * j] = src[t + 256 * j];
    if (t < TILE_KC) lds_cn[0][t] = CN[t];
  }
  __syncthreads();

  // ---- Build persistent A fragments (16x32 bf16, documented VGPR layout):
  //      lanes 0-15 : row M=lane,    K = {0..7, 16..23}
  //      lanes 16-31: row M=lane-16, K = {8..15, 24..31}
  BF16Frag a0, a1;
  {
    const unsigned short* xr = &lds_x[(wave * 16 + nlo) * DIM];
    a0.q[0] = *(const uint4*)(xr + hb * 8);           // dims base+0..7
    a0.q[1] = *(const uint4*)(xr + 16 + hb * 8);      // dims base+16..23
    a1.q[0] = *(const uint4*)(xr + 32 + hb * 8);      // dims 32..63 fragment
    a1.q[1] = *(const uint4*)(xr + 48 + hb * 8);
  }

  unsigned mink[8];
#pragma unroll
  for (int j = 0; j < 8; ++j) mink[j] = 0xFFFFFFFFu;

  // ---- Sweep all K codes; double-buffered chunks, pipelined sub-tiles ----
  const int nchunks = K / TILE_KC;
  for (int ch = 0; ch < nchunks; ++ch) {
    const int cur = ch & 1;
    const int kc  = ch * TILE_KC;

    // Stage next chunk into the other buffer; overlaps the WMMA work below.
    if (ch + 1 < nchunks) {
      const uint4* src = (const uint4*)(CBh + (size_t)(kc + TILE_KC) * DIM);
      uint4* dst = (uint4*)lds_cb[cur ^ 1];
#pragma unroll
      for (int j = 0; j < 4; ++j) dst[t + 256 * j] = src[t + 256 * j];
      if (t < TILE_KC) lds_cn[cur ^ 1][t] = CN[kc + TILE_KC + t];
    }

    const unsigned short* cb = lds_cb[cur];
    const float* cna = lds_cn[cur];

    // Pipeline: load B(sub) -> min-update(acc[sub-1]) -> wmma(sub).
    // The VALU min-update hides B-load latency and fills the
    // WMMA->VALU hazard window (co-executes with the matrix pipe).
    BPipe p[2];
    v8f   acc[2];
    load_b(cb, cna, 0, nlo, hb, p[0]);
    acc[0] = wmma2(a0, a1, p[0]);
#pragma unroll
    for (int sub = 1; sub < TILE_KC / 16; ++sub) {
      const int cs = sub & 1, ps = cs ^ 1;
      load_b(cb, cna, sub, nlo, hb, p[cs]);
      min_update(acc[ps], p[ps].cn,
                 (unsigned)(kc + (sub - 1) * 16 + nlo), mink);
      acc[cs] = wmma2(a0, a1, p[cs]);
    }
    min_update(acc[1], p[1].cn,
               (unsigned)(kc + (TILE_KC / 16 - 1) * 16 + nlo), mink);

    __syncthreads();  // next buffer staged + current buffer reads done
  }

  // ---- Reduce argmin across the 16 columns (xor tree stays in half-group) --
#pragma unroll
  for (int off = 1; off < 16; off <<= 1) {
#pragma unroll
    for (int j = 0; j < 8; ++j) {
      unsigned ok = (unsigned)__shfl_xor((int)mink[j], off, 32);
      mink[j] = mink[j] < ok ? mink[j] : ok;
    }
  }
  if (nlo == 0) {
#pragma unroll
    for (int j = 0; j < 8; ++j)
      lds_idx[wave * 16 + hb * 8 + j] = (int)(mink[j] & IDX_MASK);
  }
  __syncthreads();

  // ---- Gather: out[0] = z + (c - z)  (faithful rounding), out[1] = c ------
  {
    const float* zb = X   + (size_t)(block_row + wave * 16) * DIM;
    float*       o0 = out + (size_t)(block_row + wave * 16) * DIM;
    float*       o1 = o0  + (size_t)nrows * DIM;
#pragma unroll
    for (int i = 0; i < 8; ++i) {
      int r   = i * 2 + hb;          // row within wave tile, 0..15
      int col = nlo * 4;
      int idx = lds_idx[wave * 16 + r];
      float4 c = *(const float4*)(CB + (size_t)idx * DIM + col);
      float4 z = *(const float4*)(zb + r * DIM + col);
      float4 st;
      st.x = z.x + (c.x - z.x);
      st.y = z.y + (c.y - z.y);
      st.z = z.z + (c.z - z.z);
      st.w = z.w + (c.w - z.w);
      *(float4*)(o0 + r * DIM + col) = st;
      *(float4*)(o1 + r * DIM + col) = c;
    }
  }
}

// ---------------------------------------------------------------------------
extern "C" void kernel_launch(void* const* d_in, const int* in_sizes, int n_in,
                              void* d_out, int out_size, void* d_ws, size_t ws_size,
                              hipStream_t stream) {
  const float* X  = (const float*)d_in[0];   // z_e_x   [N, 64] f32
  const float* CB = (const float*)d_in[1];   // codebook [K, 64] f32
  float* out = (float*)d_out;                // [2, N, 64] f32

  const int nrows = in_sizes[0] / DIM;       // 131072
  const int K     = in_sizes[1] / DIM;       // 4096

  unsigned short* CBh = (unsigned short*)d_ws;                      // K*64 bf16
  float* CN = (float*)((char*)d_ws + (size_t)K * DIM * sizeof(unsigned short));

  vq_setup_kernel<<<(K + 255) / 256, 256, 0, stream>>>(CB, CBh, CN, K);
  vq_main_kernel<<<nrows / TILE_M, 256, 0, stream>>>(X, CB, CBh, CN, out,
                                                     nrows, K);
}
